// QuantumGeometricAttention_65455301591553
// MI455X (gfx1250) — compile-verified, matrix operand order checked by reading, not run on hardware
//
#include <hip/hip_runtime.h>
#include <hip/hip_bf16.h>

// ---------------------------------------------------------------------------
// QuantumGeometricAttention for MI455X (gfx1250, wave32, WMMA).
// bf16 WMMA (f32 accum) GEMMs, complex ops as real GEMM combos,
// metric+flow+invproj fused into one [M,H] matrix:
//   out = Re(q_attn) * (g * Mf^10 * Wp_inv^T) + bp_inv,  Mf = 0.9 I + 0.1 Fw.
// GEMM kernel: 128x128 block tile, 8 waves, each wave 32x64 (2x4 16x16 WMMA
// frags -> 8 WMMA per k-slice), ping-pong LDS + register pipeline,
// vector bf16 conversion, prefetch.
// ---------------------------------------------------------------------------

typedef __attribute__((ext_vector_type(16))) __bf16 v16bf;
typedef __attribute__((ext_vector_type(8)))  __bf16 v8bf;
typedef __attribute__((ext_vector_type(4)))  __bf16 v4bf;
typedef __attribute__((ext_vector_type(8)))  float  v8f;
typedef __attribute__((ext_vector_type(4)))  float  vf4;

#define TM 128
#define TN 128
#define TKK 32
#define LDSK 40    // TKK + 8 pad -> 80B row stride (16B-aligned fragment loads)

union FragBF { v16bf v; v8bf h[2]; };

static __device__ __forceinline__ v4bf cvt4(float x, float y, float z, float w) {
    vf4 f; f.x = x; f.y = y; f.z = z; f.w = w;
    return __builtin_convertvector(f, v4bf);
}

// C[M,N] = alpha * A[M,K] x op(B) (+ C if BETA) (+ bias if HAS_BIAS),
// batched over blockIdx.z.
// B_IS_NT=true : B is [N,K] row-major; false: B is [K,N] row-major.
template<bool B_IS_NT, bool BETA, bool HAS_BIAS>
__global__ __launch_bounds__(256) void gemm_wmma(
    const float* __restrict__ A, const float* __restrict__ B,
    const float* __restrict__ bias, float* __restrict__ C,
    int M, int N, int K, long sA, long sB, long sC, float alpha)
{
    __shared__ __bf16 As[2][TM * LDSK];
    __shared__ __bf16 Bs[2][TN * LDSK];

    const int bz = blockIdx.z;
    A += (long)bz * sA;  B += (long)bz * sB;  C += (long)bz * sC;

    const int tile_n = blockIdx.x * TN;
    const int tile_m = blockIdx.y * TM;
    const int tid  = threadIdx.x;
    const int lane = tid & 31;
    const int wave = tid >> 5;       // 8 waves
    const int wm   = wave & 3;       // 4 along M (32 rows each)
    const int wn   = wave >> 2;      // 2 along N (64 cols each)
    const int l16  = lane & 15;
    const int half = lane >> 4;      // K group {0..7,16..23} vs {8..15,24..31}
    const int khi  = half * 8;

    // per-thread global-load slots
    const int ldr = tid >> 3;        // 0..31
    const int ldk = (tid & 7) * 4;   // 0,4,..,28
    const int n4  = (tid & 31) * 4;  // NN: 0..124
    const int kg  = (tid >> 5) * 4;  // NN: k rows kg..kg+3 (0,4,..,28)

    vf4 ra[4], rb[4];

    auto loadG = [&](int k0) {
#pragma unroll
        for (int p = 0; p < 4; ++p)
            ra[p] = *(const vf4*)(A + (tile_m + ldr + 32 * p) * K + k0 + ldk);
        if (B_IS_NT) {
#pragma unroll
            for (int p = 0; p < 4; ++p)
                rb[p] = *(const vf4*)(B + (tile_n + ldr + 32 * p) * K + k0 + ldk);
        } else {
#pragma unroll
            for (int p = 0; p < 4; ++p)
                rb[p] = *(const vf4*)(B + (k0 + kg + p) * N + tile_n + n4);
        }
    };
    auto storeL = [&](int buf) {
#pragma unroll
        for (int p = 0; p < 4; ++p)
            *(v4bf*)&As[buf][(ldr + 32 * p) * LDSK + ldk] =
                __builtin_convertvector(ra[p], v4bf);
        if (B_IS_NT) {
#pragma unroll
            for (int p = 0; p < 4; ++p)
                *(v4bf*)&Bs[buf][(ldr + 32 * p) * LDSK + ldk] =
                    __builtin_convertvector(rb[p], v4bf);
        } else {
            // transpose 4x4 in registers: cols n4..n4+3 each get k rows kg..kg+3
            *(v4bf*)&Bs[buf][(n4 + 0) * LDSK + kg] = cvt4(rb[0].x, rb[1].x, rb[2].x, rb[3].x);
            *(v4bf*)&Bs[buf][(n4 + 1) * LDSK + kg] = cvt4(rb[0].y, rb[1].y, rb[2].y, rb[3].y);
            *(v4bf*)&Bs[buf][(n4 + 2) * LDSK + kg] = cvt4(rb[0].z, rb[1].z, rb[2].z, rb[3].z);
            *(v4bf*)&Bs[buf][(n4 + 3) * LDSK + kg] = cvt4(rb[0].w, rb[1].w, rb[2].w, rb[3].w);
        }
    };

    v8f acc[2][4] = {};

    loadG(0);
    storeL(0);
    __syncthreads();

    const int nk = K / TKK;
    for (int t = 0; t < nk; ++t) {
        const int cur = t & 1;
        if (t + 1 < nk) {
            loadG((t + 1) * TKK);
            if (t + 2 < nk) {   // warm L2 for tile t+2 (global_prefetch_b8)
                __builtin_prefetch(A + (tile_m + ldr) * K + (t + 2) * TKK + ldk, 0, 3);
                if (B_IS_NT)
                    __builtin_prefetch(B + (tile_n + ldr) * K + (t + 2) * TKK + ldk, 0, 3);
                else
                    __builtin_prefetch(B + ((t + 2) * TKK + kg) * N + tile_n + n4, 0, 3);
            }
        }

        FragBF fa[2], fb[4];
#pragma unroll
        for (int i = 0; i < 2; ++i) {
            const __bf16* ap = &As[cur][(wm * 32 + i * 16 + l16) * LDSK];
            fa[i].h[0] = *(const v8bf*)(ap + khi);       // K khi..khi+7
            fa[i].h[1] = *(const v8bf*)(ap + khi + 16);  // K khi+16..khi+23
        }
#pragma unroll
        for (int j = 0; j < 4; ++j) {
            const __bf16* bp_ = &Bs[cur][(wn * 64 + j * 16 + l16) * LDSK];
            fb[j].h[0] = *(const v8bf*)(bp_ + khi);
            fb[j].h[1] = *(const v8bf*)(bp_ + khi + 16);
        }
#pragma unroll
        for (int i = 0; i < 2; ++i)
#pragma unroll
            for (int j = 0; j < 4; ++j)
                acc[i][j] = __builtin_amdgcn_wmma_f32_16x16x32_bf16(
                    false, fa[i].v, false, fb[j].v,
                    (short)0, acc[i][j], false, false);

        if (t + 1 < nk) storeL(cur ^ 1);
        __syncthreads();
    }

    // epilogue: C VGPR r -> M = r + 8*half ; N = l16 (+16 per j)
    const int gc  = tile_n + wn * 64 + l16;
    const int gr0 = tile_m + wm * 32 + half * 8;
    float bj[4] = {0.f, 0.f, 0.f, 0.f};
    if (HAS_BIAS) {
#pragma unroll
        for (int j = 0; j < 4; ++j) bj[j] = bias[gc + j * 16];
    }
#pragma unroll
    for (int i = 0; i < 2; ++i)
#pragma unroll
        for (int r = 0; r < 8; ++r) {
            const int off = (gr0 + i * 16 + r) * N + gc;
#pragma unroll
            for (int j = 0; j < 4; ++j) {
                float v = alpha * acc[i][j][r] + bj[j];
                if (BETA) v += C[off + j * 16];
                C[off + j * 16] = v;
            }
        }
}

// complex softmax over rows of [rows x cols] (cols = 2048 = 8*256), in place.
__global__ __launch_bounds__(256) void complex_softmax_kernel(
    float* __restrict__ sr, float* __restrict__ si, int cols)
{
    const int P = 8;
    const long row = blockIdx.x;
    float* pr = sr + row * (long)cols;
    float* pi = si + row * (long)cols;
    const int tid = threadIdx.x, lane = tid & 31, wave = tid >> 5;
    __shared__ float red[8];

    float vr[P], vi[P], va[P], ve[P];
    float mx = -3.4e38f;
#pragma unroll
    for (int p = 0; p < P; ++p) {
        const int c = tid + p * 256;
        vr[p] = pr[c]; vi[p] = pi[c];
        va[p] = sqrtf(vr[p] * vr[p] + vi[p] * vi[p]);
        mx = fmaxf(mx, va[p]);
    }
    for (int o = 16; o > 0; o >>= 1) mx = fmaxf(mx, __shfl_down(mx, o, 32));
    if (lane == 0) red[wave] = mx;
    __syncthreads();
    if (tid == 0) { float m = red[0]; for (int w = 1; w < 8; ++w) m = fmaxf(m, red[w]); red[0] = m; }
    __syncthreads();
    mx = red[0];
    __syncthreads();

    float s = 0.f;
#pragma unroll
    for (int p = 0; p < P; ++p) { ve[p] = __expf(va[p] - mx); s += ve[p]; }
    for (int o = 16; o > 0; o >>= 1) s += __shfl_down(s, o, 32);
    if (lane == 0) red[wave] = s;
    __syncthreads();
    if (tid == 0) { float t = 0.f; for (int w = 0; w < 8; ++w) t += red[w]; red[0] = t; }
    __syncthreads();
    s = red[0];

    const float inv = 1.f / (s + 1e-8f);
#pragma unroll
    for (int p = 0; p < P; ++p) {
        const int c = tid + p * 256;
        const float f = ve[p] * inv / (va[p] + 1e-8f);
        pr[c] = vr[p] * f;
        pi[c] = vi[p] * f;
    }
}

// Mf = 0.9*I + 0.1*Fw
__global__ void build_flow(const float* __restrict__ Fw, float* __restrict__ Mf, int n)
{
    const long idx = (long)blockIdx.x * blockDim.x + threadIdx.x;
    if (idx < (long)n * n) {
        const int i = (int)(idx / n), j = (int)(idx % n);
        Mf[idx] = 0.1f * Fw[idx] + (i == j ? 0.9f : 0.f);
    }
}

static inline void G(bool b_nt, const float* A, const float* B, const float* bias,
                     float* C, long M, long N, long K, long sA, long sB, long sC,
                     float alpha, int beta, int batch, hipStream_t stream)
{
    dim3 grid((unsigned)(N / TN), (unsigned)(M / TM), (unsigned)batch), blk(256);
#define LG(NT, BT, BI) gemm_wmma<NT, BT, BI><<<grid, blk, 0, stream>>>( \
        A, B, bias, C, (int)M, (int)N, (int)K, sA, sB, sC, alpha)
    if (b_nt) {
        if (beta) { if (bias) LG(true,  true,  true); else LG(true,  true,  false); }
        else      { if (bias) LG(true,  false, true); else LG(true,  false, false); }
    } else {
        if (beta) { if (bias) LG(false, true,  true); else LG(false, true,  false); }
        else      { if (bias) LG(false, false, true); else LG(false, false, false); }
    }
#undef LG
}

extern "C" void kernel_launch(void* const* d_in, const int* in_sizes, int n_in,
                              void* d_out, int out_size, void* d_ws, size_t ws_size,
                              hipStream_t stream)
{
    (void)in_sizes; (void)n_in; (void)out_size; (void)ws_size;
    const float* x      = (const float*)d_in[0];
    const float* Wp     = (const float*)d_in[1];
    const float* bp     = (const float*)d_in[2];
    const float* Wp_inv = (const float*)d_in[3];
    const float* bp_inv = (const float*)d_in[4];
    const float* aWr    = (const float*)d_in[5];
    const float* aWi    = (const float*)d_in[6];
    const float* abr    = (const float*)d_in[7];
    const float* abi    = (const float*)d_in[8];
    const float* metric = (const float*)d_in[9];
    const float* flow_W = (const float*)d_in[10];
    float* out = (float*)d_out;

    const long Mm = 1024, H = 1024, S = 2048, Bb = 2, BS = Bb * S;

    // workspace layout (~148 MB fp32)
    float* w  = (float*)d_ws;
    float* qr = w; w += BS * Mm;
    float* qi = w; w += BS * Mm;
    float* ar = w; w += BS * Mm;
    float* ai = w; w += BS * Mm;
    float* sr = w; w += Bb * S * S;
    float* si = w; w += Bb * S * S;
    float* t0 = w; w += Mm * Mm;   // Mf
    float* t1 = w; w += Mm * Mm;   // Mf^2
    float* t2 = w; w += Mm * Mm;   // Mf^4 -> g*Mf^10
    float* t3 = w; w += Mm * Mm;   // Mf^8 -> T
    float* t4 = w; w += Mm * Mm;   // Mf^10

    const float scale = 0.03125f;  // 1/sqrt(1024)

    // 1) h = x @ Wp^T + bp -> qr (q imaginary part implicitly zero)
    G(true, x, Wp, bp, qr, BS, Mm, H, 0, 0, 0, 1.f, 0, 1, stream);

    // 2) three complex linear-attention layers
    for (int l = 0; l < 3; ++l) {
        const float* Wr = aWr + (long)l * Mm * Mm;
        const float* Wi = aWi + (long)l * Mm * Mm;
        const float* br = abr + (long)l * Mm;
        const float* bi = abi + (long)l * Mm;
        // a = q @ W^T + b
        G(true, qr, Wr, br, ar, BS, Mm, Mm, 0, 0, 0, 1.f, 0, 1, stream);
        G(true, qr, Wi, bi, ai, BS, Mm, Mm, 0, 0, 0, 1.f, 0, 1, stream);
        if (l > 0) {
            G(true, qi, Wi, nullptr, ar, BS, Mm, Mm, 0, 0, 0, -1.f, 1, 1, stream);
            G(true, qi, Wr, nullptr, ai, BS, Mm, Mm, 0, 0, 0,  1.f, 1, 1, stream);
        }
        // scores = scale * a @ conj(a)^T   (per batch via grid.z)
        G(true, ar, ar, nullptr, sr, S, S, Mm, S*Mm, S*Mm, S*S,  scale, 0, 2, stream);
        G(true, ai, ai, nullptr, sr, S, S, Mm, S*Mm, S*Mm, S*S,  scale, 1, 2, stream);
        G(true, ai, ar, nullptr, si, S, S, Mm, S*Mm, S*Mm, S*S,  scale, 0, 2, stream);
        G(true, ar, ai, nullptr, si, S, S, Mm, S*Mm, S*Mm, S*S, -scale, 1, 2, stream);
        // complex softmax over last axis, in place
        complex_softmax_kernel<<<dim3((unsigned)BS), dim3(256), 0, stream>>>(sr, si, (int)S);
        // q_new = attn @ q  (into a-buffers, then pointer swap)
        G(false, sr, qr, nullptr, ar, S, Mm, S, S*S, S*Mm, S*Mm, 1.f, 0, 2, stream);
        G(false, si, qr, nullptr, ai, S, Mm, S, S*S, S*Mm, S*Mm, 1.f, 0, 2, stream);
        if (l > 0) {
            G(false, si, qi, nullptr, ar, S, Mm, S, S*S, S*Mm, S*Mm, -1.f, 1, 2, stream);
            G(false, sr, qi, nullptr, ai, S, Mm, S, S*S, S*Mm, S*Mm,  1.f, 1, 2, stream);
        }
        float* t;
        t = qr; qr = ar; ar = t;
        t = qi; qi = ai; ai = t;
    }

    // 3) fused tail: out = Re(q) * (g * Mf^10 * Wp_inv^T) + bp_inv
    build_flow<<<dim3((unsigned)((Mm * Mm + 255) / 256)), dim3(256), 0, stream>>>(flow_W, t0, (int)Mm);
    G(false, t0, t0, nullptr, t1, Mm, Mm, Mm, 0, 0, 0, 1.f, 0, 1, stream);   // Mf^2
    G(false, t1, t1, nullptr, t2, Mm, Mm, Mm, 0, 0, 0, 1.f, 0, 1, stream);   // Mf^4
    G(false, t2, t2, nullptr, t3, Mm, Mm, Mm, 0, 0, 0, 1.f, 0, 1, stream);   // Mf^8
    G(false, t3, t1, nullptr, t4, Mm, Mm, Mm, 0, 0, 0, 1.f, 0, 1, stream);   // Mf^10
    G(false, metric, t4, nullptr, t2, Mm, Mm, Mm, 0, 0, 0, 1.f, 0, 1, stream);   // g*Mf^10
    G(true,  t2, Wp_inv, nullptr, t3, Mm, H, Mm, 0, 0, 0, 1.f, 0, 1, stream);    // T
    G(false, qr, t3, bp_inv, out, BS, H, Mm, 0, 0, 0, 1.f, 0, 1, stream);        // out
}
